// GATModel_37331855737170
// MI455X (gfx1250) — matrix operand order repaired and checked by reading, not compiled, not run on hardware
//
#include <hip/hip_runtime.h>

// ---------------- problem constants (match reference) ----------------
#define NN     50000
#define EE     800000
#define ETOT   (EE + NN)        // edges + self loops
#define F_IN   128
#define F_HID  256              // HEADS * HID
#define F_OUT  64
#define HEADS  4
#define NEG_SLOPE 0.2f

typedef __attribute__((ext_vector_type(16))) __bf16 v16bf;
typedef __attribute__((ext_vector_type(8)))  float  v8f;
typedef unsigned int u32x4 __attribute__((ext_vector_type(4)));

// ---------------- helpers ----------------
__device__ __forceinline__ unsigned short f2bf(float f) {
    unsigned u = __float_as_uint(f);
    unsigned r = 0x7fffu + ((u >> 16) & 1u);     // round-to-nearest-even
    return (unsigned short)((u + r) >> 16);
}
__device__ __forceinline__ unsigned keyenc(float f) {   // order-preserving f32 -> u32
    unsigned b = __float_as_uint(f);
    return (b & 0x80000000u) ? ~b : (b | 0x80000000u);
}
__device__ __forceinline__ float keydec(unsigned k) {
    return (k & 0x80000000u) ? __uint_as_float(k & 0x7fffffffu) : __uint_as_float(~k);
}
__device__ __forceinline__ float lrelu(float v) { return v > 0.f ? v : NEG_SLOPE * v; }

// ---------------- small streaming kernels ----------------
__global__ void k_zero(unsigned* __restrict__ p, int n) {
    int i = blockIdx.x * 256 + threadIdx.x;
    if (i < n) p[i] = 0u;
}
__global__ void k_f32_to_bf16(const float* __restrict__ in, unsigned short* __restrict__ out, int n) {
    int i = blockIdx.x * 256 + threadIdx.x;
    if (i < n) out[i] = f2bf(in[i]);
}
// W [K, Nout] f32 (row-major) -> Wt [Nout, K] bf16  (so B columns become contiguous)
__global__ void k_wt_bf16(const float* __restrict__ W, unsigned short* __restrict__ Wt, int K, int Nout) {
    int i = blockIdx.x * 256 + threadIdx.x;
    if (i < K * Nout) {
        int k = i / Nout, c = i - k * Nout;
        Wt[(size_t)c * K + k] = f2bf(W[i]);
    }
}

// ---------------- bf16 WMMA GEMM: C[M,Nout] = A[M,K] * Bt^T + bias ----------------
// One wave32 computes a 16x64 C strip (4 accumulators): the A fragment is loaded
// once per K-step and reused by 4 v_wmma_f32_16x16x32_bf16 issues (2.5 b128 loads
// per WMMA instead of 4, and 4x less A refetch from L2).
__global__ __launch_bounds__(128) void k_gemm_bf16(
    const unsigned short* __restrict__ A,    // [M,K] bf16 row-major
    const unsigned short* __restrict__ Bt,   // [Nout,K] bf16 (weights transposed)
    const float* __restrict__ bias,          // [Nout]
    float* __restrict__ C,                   // [M,Nout] f32
    int M, int K, int Nout)
{
    const int lane    = threadIdx.x & 31;
    const int wave    = threadIdx.x >> 5;
    const int ntileN4 = Nout >> 6;                   // 64-wide N strips
    const int tiles   = (M >> 4) * ntileN4;
    const int tile    = blockIdx.x * 4 + wave;
    if (tile >= tiles) return;
    const int tm  = tile / ntileN4;
    const int tn4 = tile - tm * ntileN4;

    const int half = lane >> 4;      // 0 / 1
    const int l15  = lane & 15;

    // A: row = l15, K-halves interleave by 8 per documented 16-bit A layout
    const unsigned short* arow = A + (size_t)(tm * 16 + l15) * K + half * 8;
    // B: col = l15 (+16 per sub-tile), K-halves interleave by 8 (same as A)
    const unsigned short* brow[4];
#pragma unroll
    for (int j = 0; j < 4; ++j)
        brow[j] = Bt + (size_t)(tn4 * 64 + j * 16 + l15) * K + half * 8;

    v8f acc[4] = {};
    for (int k0 = 0; k0 < K; k0 += 32) {
        union { u32x4 q[2]; v16bf v; } a;
        a.q[0] = *(const u32x4*)(arow + k0);        // K = k0+8h   .. +7   -> VGPR0-3
        a.q[1] = *(const u32x4*)(arow + k0 + 16);   // K = k0+16+8h.. +7   -> VGPR4-7
#pragma unroll
        for (int j = 0; j < 4; ++j) {
            union { u32x4 q[2]; v16bf v; } b;
            b.q[0] = *(const u32x4*)(brow[j] + k0);      // K = k0+8h   .. +7
            b.q[1] = *(const u32x4*)(brow[j] + k0 + 16); // K = k0+16+8h.. +7
            acc[j] = __builtin_amdgcn_wmma_f32_16x16x32_bf16(
                         false, a.v, false, b.v, (short)0, acc[j], false, false);
        }
    }

    float* cbase = C + (size_t)(tm * 16 + half * 8) * Nout;
#pragma unroll
    for (int j = 0; j < 4; ++j) {
        const int col = tn4 * 64 + j * 16 + l15;
        const float bv = bias[col];
#pragma unroll
        for (int i = 0; i < 8; ++i)                  // C VGPR i -> row (8*half + i)
            cbase[(size_t)i * Nout + col] = acc[j][i] + bv;
    }
}

// ---------------- per-node attention logits: as/ad[n,h] = <xh[n,h,:], a_src/dst[h,:]> ----
__global__ void k_node_alpha(const float* __restrict__ xh,
                             const float* __restrict__ a_src, const float* __restrict__ a_dst,
                             float* __restrict__ as_, float* __restrict__ ad_)
{
    int i = blockIdx.x * 256 + threadIdx.x;
    if (i >= NN * HEADS) return;
    int n = i >> 2, h = i & 3;
    const float* xp = xh + (size_t)n * F_HID + h * 64;
    const float* sp = a_src + h * 64;
    const float* dp = a_dst + h * 64;
    float s = 0.f, d = 0.f;
#pragma unroll 8
    for (int c = 0; c < 64; ++c) { float v = xp[c]; s += v * sp[c]; d += v * dp[c]; }
    as_[i] = s; ad_[i] = d;
}

// ---------------- edge pass 1: segment max (atomicMax on order-preserving key) --------
__global__ void k_edge_max(const int* __restrict__ ei,
                           const float* __restrict__ as_, const float* __restrict__ ad_,
                           unsigned* __restrict__ mkey)
{
    int i = blockIdx.x * 256 + threadIdx.x;
    if (i >= ETOT * HEADS) return;
    int e = i >> 2, h = i & 3;
    int s, d;
    if (e < EE) { s = ei[e]; d = ei[EE + e]; } else { s = d = e - EE; }
    float v = lrelu(as_[s * 4 + h] + ad_[d * 4 + h]);
    atomicMax(mkey + d * 4 + h, keyenc(v));
}

// ---------------- edge pass 2: denom = segment_sum(exp(e - m)) ------------------------
__global__ void k_edge_sum(const int* __restrict__ ei,
                           const float* __restrict__ as_, const float* __restrict__ ad_,
                           const unsigned* __restrict__ mkey, float* __restrict__ denom)
{
    int i = blockIdx.x * 256 + threadIdx.x;
    if (i >= ETOT * HEADS) return;
    int e = i >> 2, h = i & 3;
    int s, d;
    if (e < EE) { s = ei[e]; d = ei[EE + e]; } else { s = d = e - EE; }
    float v = lrelu(as_[s * 4 + h] + ad_[d * 4 + h]);
    atomicAdd(denom + d * 4 + h, __expf(v - keydec(mkey[d * 4 + h])));
}

// ---------------- edge pass 3: agg[dst] += alpha * xh[src]  (wave per edge) -----------
__global__ __launch_bounds__(256) void k_edge_scatter(
    const int* __restrict__ ei,
    const float* __restrict__ as_, const float* __restrict__ ad_,
    const unsigned* __restrict__ mkey, const float* __restrict__ denom,
    const float* __restrict__ xh, float* __restrict__ agg)
{
    int wid  = (blockIdx.x * 256 + threadIdx.x) >> 5;
    int lane = threadIdx.x & 31;
    if (wid >= ETOT) return;
    int s, d;
    if (wid < EE) { s = ei[wid]; d = ei[EE + wid]; } else { s = d = wid - EE; }

    const float* xs = xh + (size_t)s * F_HID;
    float*       ag = agg + (size_t)d * F_HID;
    __builtin_prefetch(xs + lane, 0, 0);             // global_prefetch of gathered row

    // lanes 0..3 carry the 4 head weights (replicated in lanes 4..31)
    int   hh = lane & 3;
    float v  = lrelu(as_[s * 4 + hh] + ad_[d * 4 + hh]);
    float wh = __expf(v - keydec(mkey[d * 4 + hh])) / (denom[d * 4 + hh] + 1e-16f);

#pragma unroll
    for (int r = 0; r < 8; ++r) {
        int c = r * 32 + lane;                       // channel 0..255, head = c>>6
        float w = __shfl(wh, c >> 6, 32);
        atomicAdd(ag + c, xs[c] * w);
    }
}

// ---------------- h = relu(agg + bias), stored as bf16 for the next GEMM --------------
__global__ void k_relu_bias_bf16(const float* __restrict__ agg, const float* __restrict__ bias,
                                 unsigned short* __restrict__ hbf, int n)
{
    int i = blockIdx.x * 256 + threadIdx.x;
    if (i < n) {
        float v = agg[i] + bias[i & (F_HID - 1)];
        hbf[i] = f2bf(v > 0.f ? v : 0.f);
    }
}

// =====================================================================================
static inline int cdiv(long long a, long long b) { return (int)((a + b - 1) / b); }

extern "C" void kernel_launch(void* const* d_in, const int* in_sizes, int n_in,
                              void* d_out, int out_size, void* d_ws, size_t ws_size,
                              hipStream_t stream)
{
    (void)in_sizes; (void)n_in; (void)out_size; (void)ws_size;
    const float* x      = (const float*)d_in[0];
    const int*   ei     = (const int*)  d_in[1];   // [2,E]
    /* d_in[2] = edge_attr: unused (edge_dim=None) */
    const float* W1     = (const float*)d_in[3];
    const float* a_src1 = (const float*)d_in[4];
    const float* a_dst1 = (const float*)d_in[5];
    const float* b1     = (const float*)d_in[6];
    const float* W2     = (const float*)d_in[7];
    const float* a_src2 = (const float*)d_in[8];
    const float* a_dst2 = (const float*)d_in[9];
    const float* b2     = (const float*)d_in[10];
    const float* Wfc    = (const float*)d_in[11];
    const float* bfc    = (const float*)d_in[12];
    float* out = (float*)d_out;

    // ---- carve workspace (~144 MB), 256B aligned ----
    char* p = (char*)d_ws;
    auto carve = [&](size_t bytes) { void* r = (void*)p; p += (bytes + 255) & ~(size_t)255; return r; };
    unsigned short* xbf   = (unsigned short*)carve((size_t)NN * F_IN  * 2);
    unsigned short* w1t   = (unsigned short*)carve((size_t)F_IN  * F_HID * 2);
    unsigned short* w2t   = (unsigned short*)carve((size_t)F_HID * F_HID * 2);
    unsigned short* wfct  = (unsigned short*)carve((size_t)F_HID * F_OUT * 2);
    float*          xh    = (float*)         carve((size_t)NN * F_HID * 4);  // GEMM output (both layers)
    float*          agg   = (float*)         carve((size_t)NN * F_HID * 4);  // scatter accumulator
    unsigned short* hbf   = (unsigned short*)carve((size_t)NN * F_HID * 2);  // relu'd features, bf16
    float*          as_   = (float*)         carve((size_t)NN * HEADS * 4);
    float*          ad_   = (float*)         carve((size_t)NN * HEADS * 4);
    unsigned*       mkey  = (unsigned*)      carve((size_t)NN * HEADS * 4);
    float*          denom = (float*)         carve((size_t)NN * HEADS * 4);

    const int NH   = NN * HEADS;                 // 200k
    const int EH   = ETOT * HEADS;               // 3.4M
    const int NF   = NN * F_HID;                 // 12.8M

    // ---- stage A: precision conversion / weight transposes ----
    k_f32_to_bf16<<<cdiv((size_t)NN * F_IN, 256), 256, 0, stream>>>(x, xbf, NN * F_IN);
    k_wt_bf16<<<cdiv(F_IN  * F_HID, 256), 256, 0, stream>>>(W1,  w1t,  F_IN,  F_HID);
    k_wt_bf16<<<cdiv(F_HID * F_HID, 256), 256, 0, stream>>>(W2,  w2t,  F_HID, F_HID);
    k_wt_bf16<<<cdiv(F_HID * F_OUT, 256), 256, 0, stream>>>(Wfc, wfct, F_HID, F_OUT);

    // ---- layer 1 ----
    {
        const int tiles = (NN / 16) * (F_HID / 64);              // 16x64 strips per wave
        k_gemm_bf16<<<cdiv(tiles, 4), 128, 0, stream>>>(xbf, w1t, b1, xh, NN, F_IN, F_HID);
        k_node_alpha<<<cdiv(NH, 256), 256, 0, stream>>>(xh, a_src1, a_dst1, as_, ad_);
        k_zero<<<cdiv(NH, 256), 256, 0, stream>>>(mkey, NH);
        k_zero<<<cdiv(NH, 256), 256, 0, stream>>>((unsigned*)denom, NH);
        k_zero<<<cdiv(NF, 256), 256, 0, stream>>>((unsigned*)agg, NF);
        k_edge_max    <<<cdiv(EH, 256),        256, 0, stream>>>(ei, as_, ad_, mkey);
        k_edge_sum    <<<cdiv(EH, 256),        256, 0, stream>>>(ei, as_, ad_, mkey, denom);
        k_edge_scatter<<<cdiv(ETOT * 32, 256), 256, 0, stream>>>(ei, as_, ad_, mkey, denom, xh, agg);
        k_relu_bias_bf16<<<cdiv(NF, 256), 256, 0, stream>>>(agg, b1, hbf, NF);
    }

    // ---- layer 2 ----
    {
        const int tiles = (NN / 16) * (F_HID / 64);
        k_gemm_bf16<<<cdiv(tiles, 4), 128, 0, stream>>>(hbf, w2t, b2, xh, NN, F_HID, F_HID);
        k_node_alpha<<<cdiv(NH, 256), 256, 0, stream>>>(xh, a_src2, a_dst2, as_, ad_);
        k_zero<<<cdiv(NH, 256), 256, 0, stream>>>(mkey, NH);
        k_zero<<<cdiv(NH, 256), 256, 0, stream>>>((unsigned*)denom, NH);
        k_zero<<<cdiv(NF, 256), 256, 0, stream>>>((unsigned*)agg, NF);
        k_edge_max    <<<cdiv(EH, 256),        256, 0, stream>>>(ei, as_, ad_, mkey);
        k_edge_sum    <<<cdiv(EH, 256),        256, 0, stream>>>(ei, as_, ad_, mkey, denom);
        k_edge_scatter<<<cdiv(ETOT * 32, 256), 256, 0, stream>>>(ei, as_, ad_, mkey, denom, xh, agg);
        k_relu_bias_bf16<<<cdiv(NF, 256), 256, 0, stream>>>(agg, b2, hbf, NF);
    }

    // ---- FC head: out = h2 @ Wfc + bfc ----
    {
        const int tiles = (NN / 16) * (F_OUT / 64);              // Nout=64 -> 1 strip wide
        k_gemm_bf16<<<cdiv(tiles, 4), 128, 0, stream>>>(hbf, wfct, bfc, out, NN, F_HID, F_OUT);
    }
}
